// _FusedTrellisExpert_678604833229
// MI455X (gfx1250) — matrix operand order repaired and checked by reading, not compiled, not run on hardware
//
#include <hip/hip_runtime.h>
#include <hip/hip_bf16.h>

// Fused SwiGLU MLP for MI455X (gfx1250, wave32, WMMA).
//   h  = silu(x @ Wg) * (x @ Wu)      (Wg|Wu fused in w_gate_up[H, 2I])
//   out = h @ Wd
// fp32 operands convert to bf16 (RNE) while staging into LDS; MACs run on
// v_wmma_f32_16x16x32_bf16 with f32 accumulation. The bf16 intermediate
// h [512 x 14336] lives in the workspace (14.7 MB).
// Block tile 128x128, BK=32; 8 waves in a 4(M) x 2(N) grid, wave tile 32x64
// -> 16 WMMAs per k-step in the gate_up kernel, 8 in the down kernel.

typedef __attribute__((ext_vector_type(16))) __bf16 v16bf;
typedef __attribute__((ext_vector_type(2)))  __bf16 v2bf;
typedef __attribute__((ext_vector_type(8)))  float  f32x8;

#define HIDDEN 4096
#define INTER  14336
#define TOKENS 512

#define BM 128
#define BN 128
#define BK 32
#define LDA (BK + 8)   // padded LDS stride (bf16 elements) -> 80B rows, 16B aligned
#define LDB (BK + 8)   // B tiles stored transposed: [BN][BK]

#if defined(__has_builtin)
#if __has_builtin(__builtin_amdgcn_cvt_pk_bf16_f32)
#define HAVE_PK_BF16 1
#endif
#if __has_builtin(__builtin_amdgcn_rcpf)
#define HAVE_RCPF 1
#endif
#endif

__device__ __forceinline__ unsigned int pack_bf16x2(float a, float b) {
#ifdef HAVE_PK_BF16
  union { v2bf v; unsigned int u; } t;
  t.v = __builtin_amdgcn_cvt_pk_bf16_f32(a, b);
  return t.u;
#else
  unsigned int ua = __float_as_uint(a);
  unsigned int ub = __float_as_uint(b);
  ua = (ua + 0x7FFFu + ((ua >> 16) & 1u)) >> 16;   // RNE
  ub = (ub + 0x7FFFu + ((ub >> 16) & 1u)) >> 16;
  return ua | (ub << 16);
#endif
}
__device__ __forceinline__ unsigned short bf16_of(float a) {
  unsigned int ua = __float_as_uint(a);
  return (unsigned short)((ua + 0x7FFFu + ((ua >> 16) & 1u)) >> 16);
}
__device__ __forceinline__ float fast_rcp(float a) {
#ifdef HAVE_RCPF
  return __builtin_amdgcn_rcpf(a);   // v_rcp_f32
#else
  return 1.0f / a;
#endif
}
__device__ __forceinline__ f32x8 zero_acc() {
  f32x8 z;
#pragma unroll
  for (int i = 0; i < 8; ++i) z[i] = 0.0f;
  return z;
}

// A fragment (16x32 bf16, ISA layout): lanes 0-15 -> M=lane, lanes 16-31 -> M=lane-16.
// VGPR 0-3 hold K = 8*half + 0..7, VGPR 4-7 hold K = 16 + 8*half + 0..7.
__device__ __forceinline__ v16bf frag_a(const unsigned short* s, int m0, int lane) {
  const int row  = m0 + (lane & 15);
  const int half = lane >> 4;
  union { uint4 u[2]; v16bf v; } t;
  t.u[0] = *(const uint4*)(s + row * LDA + 8 * half);
  t.u[1] = *(const uint4*)(s + row * LDA + 16 + 8 * half);
  return t.v;
}
// B fragment (32x16 bf16): N = lane&15; K = 8*half + 0..7 then 16 + 8*half + 0..7 (matches A).
// LDS holds B transposed [BN][BK], so one lane reads two 16-byte K runs, 32 bytes apart.
__device__ __forceinline__ v16bf frag_b(const unsigned short* s, int n0, int lane) {
  const int col   = n0 + (lane & 15);
  const int halfk = lane >> 4;
  union { uint4 u[2]; v16bf v; } t;
  const uint4* p = (const uint4*)(s + col * LDB + 8 * halfk);
  t.u[0] = p[0];
  t.u[1] = p[2];
  return t.v;
}

// ---------------------------------------------------------------------------
// Kernel 1: gate_up GEMM + SwiGLU -> bf16 h in workspace
// grid = (INTER/BN, TOKENS/BM), block = 256 (8 waves: 4 along M x 2 along N)
// ---------------------------------------------------------------------------
__global__ __launch_bounds__(256)
void swiglu_gateup_kernel(const float* __restrict__ x,
                          const float* __restrict__ wgu,
                          unsigned short* __restrict__ h) {
  __shared__ unsigned short sA [BM * LDA];
  __shared__ unsigned short sBg[BN * LDB];
  __shared__ unsigned short sBu[BN * LDB];

  const int tid  = threadIdx.x;
  const int lane = tid & 31;
  const int wave = tid >> 5;
  const int m0w  = (wave & 3) * 32;   // wave tile: 32 (M) x 64 (N)
  const int n0w  = (wave >> 2) * 64;

  const int n0 = blockIdx.x * BN;     // column block in [0, INTER)
  const int m0 = blockIdx.y * BM;     // token block

  f32x8 accg[2][4], accu[2][4];
#pragma unroll
  for (int i = 0; i < 2; ++i)
#pragma unroll
    for (int j = 0; j < 4; ++j) { accg[i][j] = zero_acc(); accu[i][j] = zero_acc(); }

  const int bn = tid & 127;           // B column staged by this thread
  const int bk = (tid >> 7) * 16;     // 16 consecutive K rows

  for (int k0 = 0; k0 < HIDDEN; k0 += BK) {
    // ---- fetch A strip: x[m0:m0+128, k0:k0+32] (fp32) ----
    float4 av[4];
#pragma unroll
    for (int i = 0; i < 4; ++i) {
      const int lin = tid + i * 256;
      const int r = lin >> 3, c = (lin & 7) * 4;
      av[i] = *(const float4*)(x + (size_t)(m0 + r) * HIDDEN + k0 + c);
    }
    // ---- fetch B strips (gate & up), K-major per thread, N-coalesced per row ----
    float gv[16], uv[16];
#pragma unroll
    for (int j = 0; j < 16; ++j) {
      const size_t rowoff = (size_t)(k0 + bk + j) * (size_t)(2 * INTER);
      gv[j] = wgu[rowoff + n0 + bn];
      uv[j] = wgu[rowoff + INTER + n0 + bn];
    }
    if (k0 + BK < HIDDEN) {   // lowers to global_prefetch_b8
      __builtin_prefetch(&wgu[(size_t)(k0 + BK + bk) * (size_t)(2 * INTER) + n0 + bn], 0, 3);
      __builtin_prefetch(&wgu[(size_t)(k0 + BK + bk) * (size_t)(2 * INTER) + INTER + n0 + bn], 0, 3);
    }

    __syncthreads();  // previous iteration's fragment reads complete

    // ---- stage into LDS as bf16 ----
#pragma unroll
    for (int i = 0; i < 4; ++i) {
      const int lin = tid + i * 256;
      const int r = lin >> 3, c = (lin & 7) * 4;
      *(uint2*)(sA + r * LDA + c) =
          make_uint2(pack_bf16x2(av[i].x, av[i].y), pack_bf16x2(av[i].z, av[i].w));
    }
#pragma unroll
    for (int half = 0; half < 2; ++half) {
      const int o = half * 8;
      *(uint4*)(sBg + bn * LDB + bk + o) =
          make_uint4(pack_bf16x2(gv[o + 0], gv[o + 1]), pack_bf16x2(gv[o + 2], gv[o + 3]),
                     pack_bf16x2(gv[o + 4], gv[o + 5]), pack_bf16x2(gv[o + 6], gv[o + 7]));
      *(uint4*)(sBu + bn * LDB + bk + o) =
          make_uint4(pack_bf16x2(uv[o + 0], uv[o + 1]), pack_bf16x2(uv[o + 2], uv[o + 3]),
                     pack_bf16x2(uv[o + 4], uv[o + 5]), pack_bf16x2(uv[o + 6], uv[o + 7]));
    }

    __syncthreads();

    // ---- WMMA: 2 A-frags shared across 8 B-frags (gate + up) ----
    v16bf a[2], bg[4], bu[4];
#pragma unroll
    for (int mi = 0; mi < 2; ++mi) a[mi] = frag_a(sA, m0w + 16 * mi, lane);
#pragma unroll
    for (int ni = 0; ni < 4; ++ni) {
      bg[ni] = frag_b(sBg, n0w + 16 * ni, lane);
      bu[ni] = frag_b(sBu, n0w + 16 * ni, lane);
    }
#pragma unroll
    for (int mi = 0; mi < 2; ++mi)
#pragma unroll
      for (int ni = 0; ni < 4; ++ni) {
        accg[mi][ni] = __builtin_amdgcn_wmma_f32_16x16x32_bf16(
            false, a[mi], false, bg[ni], (short)0, accg[mi][ni], false, false);
        accu[mi][ni] = __builtin_amdgcn_wmma_f32_16x16x32_bf16(
            false, a[mi], false, bu[ni], (short)0, accu[mi][ni], false, false);
      }
  }

  // ---- epilogue: silu(gate) * up -> bf16 h ----
  // C/D layout: VGPR v, lanes 0-15 -> M=v, lanes 16-31 -> M=v+8; N = lane&15.
#pragma unroll
  for (int mi = 0; mi < 2; ++mi)
#pragma unroll
    for (int ni = 0; ni < 4; ++ni)
#pragma unroll
      for (int v = 0; v < 8; ++v) {
        const int gm = m0 + m0w + mi * 16 + v + 8 * (lane >> 4);
        const int gn = n0 + n0w + ni * 16 + (lane & 15);
        const float g = accg[mi][ni][v];
        const float u = accu[mi][ni][v];
        const float s = g * fast_rcp(1.0f + __expf(-g));   // silu via v_exp + v_rcp
        h[(size_t)gm * INTER + gn] = bf16_of(s * u);
      }
}

// ---------------------------------------------------------------------------
// Kernel 2: down projection  out = h(bf16) @ Wd(fp32->bf16) , f32 output
// grid = (HIDDEN/BN, TOKENS/BM), block = 256
// ---------------------------------------------------------------------------
__global__ __launch_bounds__(256)
void down_proj_kernel(const unsigned short* __restrict__ h,
                      const float* __restrict__ wd,
                      float* __restrict__ out) {
  __shared__ unsigned short sA[BM * LDA];
  __shared__ unsigned short sB[BN * LDB];

  const int tid  = threadIdx.x;
  const int lane = tid & 31;
  const int wave = tid >> 5;
  const int m0w  = (wave & 3) * 32;
  const int n0w  = (wave >> 2) * 64;

  const int n0 = blockIdx.x * BN;     // column block in [0, HIDDEN)
  const int m0 = blockIdx.y * BM;

  f32x8 acc[2][4];
#pragma unroll
  for (int i = 0; i < 2; ++i)
#pragma unroll
    for (int j = 0; j < 4; ++j) acc[i][j] = zero_acc();

  const int bn = tid & 127;
  const int bk = (tid >> 7) * 16;

  for (int k0 = 0; k0 < INTER; k0 += BK) {
    // ---- fetch A strip: h[m0:m0+128, k0:k0+32] already bf16 ----
    uint4 av[2];
#pragma unroll
    for (int i = 0; i < 2; ++i) {
      const int lin = tid + i * 256;
      const int r = lin >> 2, c = (lin & 3) * 8;
      av[i] = *(const uint4*)(h + (size_t)(m0 + r) * INTER + k0 + c);
    }
    // ---- fetch B strip: wd[k0:k0+32, n0:n0+128] fp32 ----
    float bv[16];
#pragma unroll
    for (int j = 0; j < 16; ++j)
      bv[j] = wd[(size_t)(k0 + bk + j) * HIDDEN + n0 + bn];
    if (k0 + BK < INTER)
      __builtin_prefetch(&wd[(size_t)(k0 + BK + bk) * HIDDEN + n0 + bn], 0, 3);

    __syncthreads();

#pragma unroll
    for (int i = 0; i < 2; ++i) {
      const int lin = tid + i * 256;
      const int r = lin >> 2, c = (lin & 3) * 8;
      *(uint4*)(sA + r * LDA + c) = av[i];
    }
#pragma unroll
    for (int half = 0; half < 2; ++half) {
      const int o = half * 8;
      *(uint4*)(sB + bn * LDB + bk + o) =
          make_uint4(pack_bf16x2(bv[o + 0], bv[o + 1]), pack_bf16x2(bv[o + 2], bv[o + 3]),
                     pack_bf16x2(bv[o + 4], bv[o + 5]), pack_bf16x2(bv[o + 6], bv[o + 7]));
    }

    __syncthreads();

    v16bf a[2], b[4];
#pragma unroll
    for (int mi = 0; mi < 2; ++mi) a[mi] = frag_a(sA, m0w + 16 * mi, lane);
#pragma unroll
    for (int ni = 0; ni < 4; ++ni) b[ni] = frag_b(sB, n0w + 16 * ni, lane);
#pragma unroll
    for (int mi = 0; mi < 2; ++mi)
#pragma unroll
      for (int ni = 0; ni < 4; ++ni)
        acc[mi][ni] = __builtin_amdgcn_wmma_f32_16x16x32_bf16(
            false, a[mi], false, b[ni], (short)0, acc[mi][ni], false, false);
  }

#pragma unroll
  for (int mi = 0; mi < 2; ++mi)
#pragma unroll
    for (int ni = 0; ni < 4; ++ni)
#pragma unroll
      for (int v = 0; v < 8; ++v) {
        const int gm = m0 + m0w + mi * 16 + v + 8 * (lane >> 4);
        const int gn = n0 + n0w + ni * 16 + (lane & 15);
        out[(size_t)gm * HIDDEN + gn] = acc[mi][ni][v];
      }
}

// ---------------------------------------------------------------------------
extern "C" void kernel_launch(void* const* d_in, const int* in_sizes, int n_in,
                              void* d_out, int out_size, void* d_ws, size_t ws_size,
                              hipStream_t stream) {
  (void)in_sizes; (void)n_in; (void)out_size; (void)ws_size;
  const float* x   = (const float*)d_in[0];   // [512, 4096]
  const float* wgu = (const float*)d_in[1];   // [4096, 28672]
  const float* wd  = (const float*)d_in[2];   // [14336, 4096]
  float* out = (float*)d_out;                 // [512, 4096]
  unsigned short* h = (unsigned short*)d_ws;  // bf16 [512, 14336] = 14.7 MB

  dim3 block(256);
  dim3 grid1(INTER / BN, TOKENS / BM);        // 112 x 4
  swiglu_gateup_kernel<<<grid1, block, 0, stream>>>(x, wgu, h);

  dim3 grid2(HIDDEN / BN, TOKENS / BM);       // 32 x 4
  down_proj_kernel<<<grid2, block, 0, stream>>>(h, wd, out);
}
